// IRParamsNet_67164698575363
// MI455X (gfx1250) — compile-verified, matrix operand order checked by reading, not run on hardware
//
#include <hip/hip_runtime.h>
#include <hip/hip_bf16.h>

typedef __attribute__((ext_vector_type(16))) _Float16 v16h;
typedef __attribute__((ext_vector_type(8)))  float    v8f;

#define T_LEN 2048
#define BLOCK 256
#define SEG   (T_LEN / BLOCK)   // 8 steps per thread in the scan phase

#define LOG2E_F 1.44269504088896340736f
#define LN2_F   0.69314718055994530942f

// raw-hardware softplus: max(x,0) + ln(1 + 2^(-|x|*log2e))
// log arg is in (1,2] and exp arg in (-inf,0] -> no denorm/range guards needed
__device__ __forceinline__ float softplusf(float x) {
    const float e = __builtin_amdgcn_exp2f(-fabsf(x) * LOG2E_F);     // v_exp_f32
    return fmaxf(x, 0.0f) + __builtin_amdgcn_logf(1.0f + e) * LN2_F; // v_log_f32
}
// raw-hardware sigmoid: 1 / (1 + 2^(-x*log2e))
__device__ __forceinline__ float sigmoidf_(float x) {
    const float e = __builtin_amdgcn_exp2f(-x * LOG2E_F);            // v_exp_f32
    return __builtin_amdgcn_rcpf(1.0f + e);                          // v_rcp_f32
}

__global__ __launch_bounds__(BLOCK)
void ir_estim_kernel(const float* __restrict__ X,    // (B, T, 5)
                     const float* __restrict__ SC,   // (B, 4)
                     const float* __restrict__ W1,   // (12, 6)
                     const float* __restrict__ b1,   // (12)
                     const float* __restrict__ W2,   // (5, 12)
                     const float* __restrict__ b2,   // (5)
                     const float* __restrict__ Wr1,  // (1, 6)
                     const float* __restrict__ br1,  // (1)
                     const float* __restrict__ Wr2,  // (1, 1)
                     const float* __restrict__ br2,  // (1)
                     float* __restrict__ out)        // (B)
{
    __shared__ float sA0[T_LEN], sA1[T_LEN], sC0[T_LEN], sC1[T_LEN]; // affine scan elements
    __shared__ float sO[T_LEN], sI[T_LEN];                           // OCV_U and I per t
    __shared__ float scA0[BLOCK], scA1[BLOCK], scC0[BLOCK], scC1[BLOCK];
    __shared__ float redS[BLOCK], redC[BLOCK];
    __shared__ float sY1;                                            // y0 = (0, sY1)

    const int b    = blockIdx.x;
    const int tid  = threadIdx.x;
    const int lane = tid & 31;
    const int wave = tid >> 5;

    const float sc0 = SC[b * 4 + 0], sc1 = SC[b * 4 + 1];
    const float sc2 = SC[b * 4 + 2], sc3 = SC[b * 4 + 3];

    // ---------- constant WMMA operands (loop invariant) ----------
    // A1 = W1 as 16x32 f16 A-operand: lane m (<12) holds row m in K=0..5.
    v16h a1 = {};
    if (lane < 12) {
        #pragma unroll
        for (int k = 0; k < 6; ++k) a1[k] = (_Float16)W1[lane * 6 + k];
    }
    // A2 = W2 as 16x32 f16 A-operand: lane m<5 holds K=0..7; lane m+16 holds K=8..11.
    v16h a2 = {};
    if (lane < 5) {
        #pragma unroll
        for (int k = 0; k < 8; ++k) a2[k] = (_Float16)W2[lane * 12 + k];
    } else if (lane >= 16 && lane < 21) {
        const int m = lane - 16;
        #pragma unroll
        for (int k = 0; k < 4; ++k) a2[k] = (_Float16)W2[m * 12 + 8 + k];
    }
    // layer-1 bias for the D1 rows this lane owns (lanes 0-15: neurons 0-7, 16-31: 8-15)
    float bb[8];
    #pragma unroll
    for (int v = 0; v < 8; ++v)
        bb[v] = (lane < 16) ? b1[v] : ((v < 4) ? b1[8 + v] : 0.0f);

    const float LBv[5] = {0.005f, 0.025f, 0.1f, 0.0f,   0.002f};
    const float UBv[5] = {0.02f,  0.07f,  1.0f, 0.055f, 0.025f};

    const float* Xb = X + (size_t)b * T_LEN * 5;

    // ============ Phase 1: theta per timestep via WMMA (16 t per tile) ============
    for (int tile = 0; tile < 16; ++tile) {
        const int tbase = wave * 256 + tile * 16;
        const int t = tbase + lane;           // meaningful for lane < 16
        float tm = 0.f, It = 0.f, Tp = 0.f, So = 0.f, tnext = 0.f;
        if (lane < 16) {
            const float* xr = Xb + (size_t)t * 5;
            tm = xr[0]; It = xr[1]; Tp = xr[2]; So = xr[4];
            tnext = xr[(t < T_LEN - 1) ? 5 : 0];   // branch-free; dt unused at t==T-1
        }
        // B1 = X^T tile: lane n holds x[t=n][0..5] in K=0..5; lanes 16-31 zero.
        v16h bx = {};
        if (lane < 16) {
            bx[0] = (_Float16)So;  bx[1] = (_Float16)Tp;
            bx[2] = (_Float16)sc0; bx[3] = (_Float16)sc1;
            bx[4] = (_Float16)sc2; bx[5] = (_Float16)sc3;
        }
        // D1 = W1 @ X^T : lane n -> neurons 0..7 (n<16) / 8..15 (n>=16) of timestep n
        v8f d1 = {};
        d1 = __builtin_amdgcn_wmma_f32_16x16x32_f16(false, a1, false, bx,
                                                    (short)0, d1, false, false);
        // softplus(h_pre + b1) -> H^T, already in the B-operand layout for WMMA2
        v16h hm = {};
        #pragma unroll
        for (int v = 0; v < 8; ++v) hm[v] = (_Float16)softplusf(d1[v] + bb[v]);
        // D2 = W2 @ H^T : lane n<16 -> p[t=n][param 0..4]
        v8f d2 = {};
        d2 = __builtin_amdgcn_wmma_f32_16x16x32_f16(false, a2, false, hm,
                                                    (short)0, d2, false, false);

        if (lane < 16) {
            float th[5];
            #pragma unroll
            for (int v = 0; v < 5; ++v)
                th[v] = LBv[v] + (UBv[v] - LBv[v]) * sigmoidf_(0.01f * (d2[v] + b2[v]));
            const float R1 = th[0], R1C1r = th[1], OCV = th[2], mMH = th[3], KH = th[4];
            sO[t] = OCV;
            sI[t] = It;
            if (t < T_LEN - 1) {
                const float dt = tnext - tm;
                sA0[t] = 1.0f + dt * (KH * It);      // A = 1 + dt*b
                sA1[t] = 1.0f - dt * R1C1r;
                sC0[t] = dt * (KH * It * mMH);       // c = dt*a
                sC1[t] = dt * (R1C1r * R1 * It);
            }
        }
    }
    __syncthreads();

    // ---- y0 from the scalar resistance net (needs sO[0]) ----
    if (tid == 0) {
        const float I0 = Xb[1], Tp0 = Xb[2], So0 = Xb[4];
        float pre = Wr1[0]*So0 + Wr1[1]*Tp0 + Wr1[2]*sc0 + Wr1[3]*sc1
                  + Wr1[4]*sc2 + Wr1[5]*sc3 + br1[0];
        float r  = softplusf(pre) * Wr2[0] + br2[0];
        float Rs = sc2 * (1.0f + r);
        sY1 = -sO[0] - I0 * Rs;                      // U_10 (U_H0 == 0)
    }

    // ============ Phase 2: block-parallel affine scan over T-1 steps ============
    // comb((A1,c1) earlier, (A2,c2) later) = (A1*A2, A2*c1 + c2), component-wise.
    float A0 = 1.f, A1 = 1.f, C0 = 0.f, C1 = 0.f;
    const int s0 = tid * SEG;
    #pragma unroll
    for (int j = 0; j < SEG; ++j) {
        const int i = s0 + j;
        if (i < T_LEN - 1) {
            const float eA0 = sA0[i], eA1 = sA1[i], eC0 = sC0[i], eC1 = sC1[i];
            C0 = eA0 * C0 + eC0;  C1 = eA1 * C1 + eC1;
            A0 *= eA0;            A1 *= eA1;
        }
    }
    scA0[tid] = A0; scA1[tid] = A1; scC0[tid] = C0; scC1[tid] = C1;

    for (int off = 1; off < BLOCK; off <<= 1) {
        __syncthreads();
        float pA0 = 1.f, pA1 = 1.f, pC0 = 0.f, pC1 = 0.f;
        if (tid >= off) {
            pA0 = scA0[tid - off]; pA1 = scA1[tid - off];
            pC0 = scC0[tid - off]; pC1 = scC1[tid - off];
        }
        __syncthreads();
        if (tid >= off) {
            scC0[tid] = scA0[tid] * pC0 + scC0[tid];
            scC1[tid] = scA1[tid] * pC1 + scC1[tid];
            scA0[tid] *= pA0; scA1[tid] *= pA1;
        }
    }
    __syncthreads();

    // exclusive prefix for this thread's segment
    float cA0 = 1.f, cA1 = 1.f, cC0 = 0.f, cC1 = 0.f;
    if (tid > 0) {
        cA0 = scA0[tid - 1]; cA1 = scA1[tid - 1];
        cC0 = scC0[tid - 1]; cC1 = scC1[tid - 1];
    }
    const float y1 = sY1;

    float psum = 0.f, pcnt = 0.f;
    if (tid == 0) {   // term at t = 0 uses y0 directly
        const float t0 = (-sO[0] - 0.0f - y1) / sI[0];
        if (t0 == t0) { psum += t0; pcnt += 1.f; }
    }
    #pragma unroll
    for (int j = 0; j < SEG; ++j) {
        const int i = s0 + j;
        if (i < T_LEN - 1) {
            const float eA0 = sA0[i], eA1 = sA1[i], eC0 = sC0[i], eC1 = sC1[i];
            cC0 = eA0 * cC0 + eC0;  cC1 = eA1 * cC1 + eC1;
            cA0 *= eA0;             cA1 *= eA1;
            const float UH = cC0;                    // y0[0] == 0
            const float U1 = cA1 * y1 + cC1;
            const float term = (-sO[i + 1] - UH - U1) * __builtin_amdgcn_rcpf(sI[i + 1]);
            if (term == term) { psum += term; pcnt += 1.f; }   // nanmean
        }
    }

    // ---- block reduction: sum / count ----
    redS[tid] = psum; redC[tid] = pcnt;
    __syncthreads();
    for (int s = BLOCK / 2; s > 0; s >>= 1) {
        if (tid < s) { redS[tid] += redS[tid + s]; redC[tid] += redC[tid + s]; }
        __syncthreads();
    }
    if (tid == 0) out[b] = redS[0] / redC[0];   // 0/0 -> NaN matches all-NaN nanmean
}

extern "C" void kernel_launch(void* const* d_in, const int* in_sizes, int n_in,
                              void* d_out, int out_size, void* d_ws, size_t ws_size,
                              hipStream_t stream) {
    const float* X   = (const float*)d_in[0];
    const float* SC  = (const float*)d_in[1];
    const float* W1  = (const float*)d_in[2];
    const float* b1  = (const float*)d_in[3];
    const float* W2  = (const float*)d_in[4];
    const float* b2  = (const float*)d_in[5];
    const float* Wr1 = (const float*)d_in[6];
    const float* br1 = (const float*)d_in[7];
    const float* Wr2 = (const float*)d_in[8];
    const float* br2 = (const float*)d_in[9];
    float* out = (float*)d_out;

    const int B = in_sizes[1] / 4;   // SC is (B, 4)

    ir_estim_kernel<<<dim3(B), dim3(BLOCK), 0, stream>>>(
        X, SC, W1, b1, W2, b2, Wr1, br1, Wr2, br2, out);
}
